// GraphSelfAttentionLayer_54829552500977
// MI455X (gfx1250) — matrix operand order repaired and checked by reading, not compiled, not run on hardware
//
#include <hip/hip_runtime.h>
#include <hip/hip_bf16.h>

#define DEVINL __device__ __forceinline__

typedef __bf16 bf16;
typedef __attribute__((ext_vector_type(16))) bf16  v16bf;
typedef __attribute__((ext_vector_type(8)))  bf16  v8bf;
typedef __attribute__((ext_vector_type(8)))  float v8f;

#define NND   4096
#define FIN   512
#define FOUT  256
#define QKVW  768
#define HEADS 4
#define DHEAD 64
#define SCALE 0.0625f        // FOUT^-0.5
#define ALPHA 0.2f
#define NEGBIG (-9.0e15f)

// ---------------- WMMA fragment helpers (wave32, per CDNA5 ISA layouts) ----

// A fragment: 16x32 bf16, row-major source A[M x K] (ld = K elems).
// lane: row = m0 + (lane&15); elems 0..7 -> K = k0+base+0..7, elems 8..15 -> K = k0+base+16..23
// with base = (lane&16)?8:0.  Two contiguous 16B loads.
DEVINL v16bf load_a16(const bf16* A, int ld, int m0, int k0) {
  int lane = threadIdx.x & 31;
  const bf16* p = A + (size_t)(m0 + (lane & 15)) * ld + k0 + ((lane & 16) ? 8 : 0);
  union { v16bf v; v8bf h[2]; } u;
  u.h[0] = *(const v8bf*)p;
  u.h[1] = *(const v8bf*)(p + 16);
  return u.v;
}

// B fragment: 32x16 bf16 fed from Bt[N x K] row-major (ld = K elems).
// lane: col n = n0 + (lane&15); elems 0..7 -> K = k0+base+0..7, elems 8..15 -> K = k0+base+16..23, base = (lane&16)?8:0.
DEVINL v16bf load_b16(const bf16* Bt, int ld, int n0, int k0) {
  int lane = threadIdx.x & 31;
  const bf16* p = Bt + (size_t)(n0 + (lane & 15)) * ld + k0 + ((lane & 16) ? 8 : 0);
  union { v16bf v; v8bf h[2]; } u;
  u.h[0] = *(const v8bf*)p;
  u.h[1] = *(const v8bf*)(p + 16);
  return u.v;
}

// A fragment from a 16x32 bf16 row-major LDS tile.
DEVINL v16bf lds_a16(const bf16* t) {
  int lane = threadIdx.x & 31;
  const bf16* p = t + (lane & 15) * 32 + ((lane & 16) ? 8 : 0);
  union { v16bf v; v8bf h[2]; } u;
  u.h[0] = *(const v8bf*)p;
  u.h[1] = *(const v8bf*)(p + 16);
  return u.v;
}

DEVINL v8f wmma_bf16(v16bf a, v16bf b, v8f c) {
  return __builtin_amdgcn_wmma_f32_16x16x32_bf16(false, a, false, b, (short)0, c,
                                                 false, false);
}

DEVINL v8f zero8() { v8f z = {0.f,0.f,0.f,0.f,0.f,0.f,0.f,0.f}; return z; }

// half-wave (16-lane) reductions; C/D layout keeps one matrix row in one half.
DEVINL float hmax16(float v) {
  #pragma unroll
  for (int m = 1; m < 16; m <<= 1) v = fmaxf(v, __shfl_xor(v, m));
  return v;
}
DEVINL float hsum16(float v) {
  #pragma unroll
  for (int m = 1; m < 16; m <<= 1) v += __shfl_xor(v, m);
  return v;
}

// ---------------- elementwise prep kernels --------------------------------

__global__ void k_cvt(const float* s, bf16* d, int n) {
  int i = blockIdx.x * 256 + threadIdx.x;
  if (i < n) d[i] = (bf16)s[i];
}

__global__ void k_tcvt(const float* s, bf16* d, int R, int C) {
  int i = blockIdx.x * 256 + threadIdx.x;
  if (i < R * C) {
    int r = i / C, c = i % C;
    d[(size_t)c * R + r] = (bf16)s[i];
  }
}

// vt[cd][j] = qkv[j][512+cd]  (V transposed, bf16), cd in [0,256)
__global__ void k_vtk(const float* qkv, bf16* vt) {
  int i = blockIdx.x * 256 + threadIdx.x;
  int j = i >> 8, cd = i & 255;
  vt[(size_t)cd * NND + j] = (bf16)qkv[(size_t)j * QKVW + 512 + cd];
}

// ---------------- generic WMMA GEMM: C[M,N] = A[M,K] * Bt[N,K]^T + bias ----

__global__ void k_gemm(const bf16* __restrict__ A, const bf16* __restrict__ Bt,
                       const float* __restrict__ bias, float* __restrict__ C,
                       int M, int Ntiles, int K) {
  int wid = threadIdx.x >> 5, lane = threadIdx.x & 31;
  int tile = blockIdx.x * 4 + wid;
  int mt = tile / Ntiles, nt = tile % Ntiles;
  int m0 = mt * 16, n0 = nt * 16;
  v8f acc = zero8();
  for (int k0 = 0; k0 < K; k0 += 32) {
    v16bf a = load_a16(A, K, m0, k0);
    v16bf b = load_b16(Bt, K, n0, k0);
    acc = wmma_bf16(a, b, acc);
  }
  int N = Ntiles * 16;
  int col = n0 + (lane & 15);
  float bv = bias ? bias[col] : 0.f;
  int rb = m0 + ((lane & 16) ? 8 : 0);
  #pragma unroll
  for (int r = 0; r < 8; r++)
    C[(size_t)(rb + r) * N + col] = acc[r] + bv;
}

// ---------------- pass 1: per-head softmax stats of Q K^T * scale ---------

__global__ void k_stats1(const bf16* __restrict__ qkvb, float* m1, float* l1) {
  int wid = threadIdx.x >> 5, lane = threadIdx.x & 31;
  int w = blockIdx.x * 4 + wid;
  int it = w & 255, h = w >> 8;
  int i0 = it * 16;
  v16bf aq0 = load_a16(qkvb + h * DHEAD, QKVW, i0, 0);
  v16bf aq1 = load_a16(qkvb + h * DHEAD, QKVW, i0, 32);
  const bf16* kb = qkvb + FOUT + h * DHEAD;
  float mr[8], lr[8];
  #pragma unroll
  for (int r = 0; r < 8; r++) { mr[r] = -3.0e38f; lr[r] = 0.f; }
  for (int j0 = 0; j0 < NND; j0 += 16) {
    v16bf b0 = load_b16(kb, QKVW, j0, 0);
    v16bf b1 = load_b16(kb, QKVW, j0, 32);
    v8f s = zero8();
    s = wmma_bf16(aq0, b0, s);
    s = wmma_bf16(aq1, b1, s);
    #pragma unroll
    for (int r = 0; r < 8; r++) {
      float x = s[r] * SCALE;
      float mn = fmaxf(mr[r], hmax16(x));
      float ts = hsum16(__expf(x - mn));
      lr[r] = lr[r] * __expf(mr[r] - mn) + ts;
      mr[r] = mn;
    }
  }
  if ((lane & 15) == 0) {
    int rb = i0 + ((lane & 16) ? 8 : 0);
    #pragma unroll
    for (int r = 0; r < 8; r++) {
      m1[h * NND + rb + r] = mr[r];
      l1[h * NND + rb + r] = lr[r];
    }
  }
}

// ---------------- pass 2: Wh0 = dots @ V (exact stats, no rescale) --------

__global__ void k_wh0(const bf16* __restrict__ qkvb, const bf16* __restrict__ vt,
                      const float* __restrict__ m1, const float* __restrict__ l1,
                      float* __restrict__ wh0) {
  __shared__ __align__(32) bf16 pt[4][16 * 32];
  int wid = threadIdx.x >> 5, lane = threadIdx.x & 31;
  int w = blockIdx.x * 4 + wid;
  int it = w & 255, h = w >> 8;
  int i0 = it * 16;
  v16bf aq0 = load_a16(qkvb + h * DHEAD, QKVW, i0, 0);
  v16bf aq1 = load_a16(qkvb + h * DHEAD, QKVW, i0, 32);
  const bf16* kb = qkvb + FOUT + h * DHEAD;
  int rboff = (lane & 16) ? 8 : 0;
  float mr[8], il[8];
  #pragma unroll
  for (int r = 0; r < 8; r++) {
    mr[r] = m1[h * NND + i0 + rboff + r];
    il[r] = 1.0f / l1[h * NND + i0 + rboff + r];
  }
  v8f o0 = zero8(), o1 = zero8(), o2 = zero8(), o3 = zero8();
  bf16* myp = pt[wid];
  const bf16* vb = vt + (size_t)h * DHEAD * NND;
  for (int j0 = 0; j0 < NND; j0 += 32) {
    #pragma unroll
    for (int sub = 0; sub < 2; sub++) {
      int jj = j0 + sub * 16;
      v16bf b0 = load_b16(kb, QKVW, jj, 0);
      v16bf b1 = load_b16(kb, QKVW, jj, 32);
      v8f s = zero8();
      s = wmma_bf16(aq0, b0, s);
      s = wmma_bf16(aq1, b1, s);
      int c = (lane & 15) + sub * 16;
      #pragma unroll
      for (int r = 0; r < 8; r++) {
        float p = __expf(s[r] * SCALE - mr[r]) * il[r];
        myp[(rboff + r) * 32 + c] = (bf16)p;
      }
    }
    asm volatile("s_wait_dscnt 0" ::: "memory");   // LDS store->load, same wave
    v16bf pa = lds_a16(myp);
    o0 = wmma_bf16(pa, load_b16(vb, NND,  0, j0), o0);
    o1 = wmma_bf16(pa, load_b16(vb, NND, 16, j0), o1);
    o2 = wmma_bf16(pa, load_b16(vb, NND, 32, j0), o2);
    o3 = wmma_bf16(pa, load_b16(vb, NND, 48, j0), o3);
  }
  int cw = lane & 15;
  #pragma unroll
  for (int r = 0; r < 8; r++) {
    size_t row = (size_t)(i0 + rboff + r) * FOUT + h * DHEAD;
    wh0[row +  0 + cw] = o0[r];
    wh0[row + 16 + cw] = o1[r];
    wh0[row + 32 + cw] = o2[r];
    wh0[row + 48 + cw] = o3[r];
  }
}

// ---------------- row softmax over 256 columns ----------------------------

__global__ void k_softmax256(const float* in, float* out) {
  __shared__ float red[256];
  int row = blockIdx.x, t = threadIdx.x;
  float x = in[(size_t)row * 256 + t];
  red[t] = x; __syncthreads();
  for (int s = 128; s > 0; s >>= 1) { if (t < s) red[t] = fmaxf(red[t], red[t + s]); __syncthreads(); }
  float m = red[0]; __syncthreads();
  float e = __expf(x - m);
  red[t] = e; __syncthreads();
  for (int s = 128; s > 0; s >>= 1) { if (t < s) red[t] += red[t + s]; __syncthreads(); }
  out[(size_t)row * 256 + t] = e / red[0];
}

// ---------------- gemv: f1 = Wh@a1, f2 = Wh@a2 ----------------------------

__global__ void k_gemv2(const float* wh, const float* a, float* f1, float* f2) {
  int wid = threadIdx.x >> 5, lane = threadIdx.x & 31;
  int row = blockIdx.x * 8 + wid;
  float s1 = 0.f, s2 = 0.f;
  for (int c = lane; c < FOUT; c += 32) {
    float wv = wh[(size_t)row * FOUT + c];
    s1 += wv * a[c];
    s2 += wv * a[FOUT + c];
  }
  #pragma unroll
  for (int m = 1; m < 32; m <<= 1) { s1 += __shfl_xor(s1, m); s2 += __shfl_xor(s2, m); }
  if (lane == 0) { f1[row] = s1; f2[row] = s2; }
}

// ---------------- pass 3: masked graph-softmax stats (j-split) ------------

__global__ void k_stats2(const bf16* __restrict__ qkvb,
                         const float* __restrict__ m1, const float* __restrict__ l1,
                         const float* __restrict__ f1, const float* __restrict__ f2,
                         const int* __restrict__ adj,
                         float* m2p, float* l2p) {
  int wid = threadIdx.x >> 5, lane = threadIdx.x & 31;
  int w = blockIdx.x * 4 + wid;
  int it = w & 255, jc = w >> 8;           // 8 j-chunks of 512
  int i0 = it * 16, rboff = (lane & 16) ? 8 : 0;
  v16bf aq[HEADS][2];
  float mh[HEADS][8], ih[HEADS][8];
  #pragma unroll
  for (int h = 0; h < HEADS; h++) {
    aq[h][0] = load_a16(qkvb + h * DHEAD, QKVW, i0, 0);
    aq[h][1] = load_a16(qkvb + h * DHEAD, QKVW, i0, 32);
    #pragma unroll
    for (int r = 0; r < 8; r++) {
      mh[h][r] = m1[h * NND + i0 + rboff + r];
      ih[h][r] = 1.0f / l1[h * NND + i0 + rboff + r];
    }
  }
  float fr[8], mr[8], lr[8];
  #pragma unroll
  for (int r = 0; r < 8; r++) { fr[r] = f1[i0 + rboff + r]; mr[r] = -3.0e38f; lr[r] = 0.f; }
  for (int j0 = jc * 512; j0 < jc * 512 + 512; j0 += 16) {
    v8f d = zero8();
    #pragma unroll
    for (int h = 0; h < HEADS; h++) {
      v16bf b0 = load_b16(qkvb + FOUT + h * DHEAD, QKVW, j0, 0);
      v16bf b1 = load_b16(qkvb + FOUT + h * DHEAD, QKVW, j0, 32);
      v8f s = zero8();
      s = wmma_bf16(aq[h][0], b0, s);
      s = wmma_bf16(aq[h][1], b1, s);
      #pragma unroll
      for (int r = 0; r < 8; r++) d[r] += __expf(s[r] * SCALE - mh[h][r]) * ih[h][r];
    }
    int j = j0 + (lane & 15);
    float f2j = f2[j];
    #pragma unroll
    for (int r = 0; r < 8; r++) {
      float e = fr[r] + f2j; e = e > 0.f ? e : ALPHA * e;
      int ad = adj[(size_t)(i0 + rboff + r) * NND + j];
      float t = ad > 0 ? (e + d[r]) : NEGBIG;
      float mn = fmaxf(mr[r], hmax16(t));
      float ts = hsum16(__expf(t - mn));
      lr[r] = lr[r] * __expf(mr[r] - mn) + ts;
      mr[r] = mn;
    }
  }
  if ((lane & 15) == 0) {
    #pragma unroll
    for (int r = 0; r < 8; r++) {
      m2p[jc * NND + i0 + rboff + r] = mr[r];
      l2p[jc * NND + i0 + rboff + r] = lr[r];
    }
  }
}

__global__ void k_comb2(const float* m2p, const float* l2p, float* m2, float* l2) {
  int i = blockIdx.x * 256 + threadIdx.x;
  float m = -3.0e38f;
  for (int c = 0; c < 8; c++) m = fmaxf(m, m2p[c * NND + i]);
  float l = 0.f;
  for (int c = 0; c < 8; c++) l += l2p[c * NND + i] * __expf(m2p[c * NND + i] - m);
  m2[i] = m; l2[i] = l;
}

// ---------------- pass 4: h' += attention @ Wh (exact stats, atomics) -----

__global__ void k_passB(const bf16* __restrict__ qkvb,
                        const float* __restrict__ m1, const float* __restrict__ l1,
                        const float* __restrict__ f1, const float* __restrict__ f2,
                        const int* __restrict__ adj,
                        const float* __restrict__ m2, const float* __restrict__ l2,
                        const bf16* __restrict__ wht, float* __restrict__ hp) {
  __shared__ __align__(32) bf16 pt[4][16 * 32];
  int wid = threadIdx.x >> 5, lane = threadIdx.x & 31;
  int w = blockIdx.x * 4 + wid;
  int it = w & 255, cs = (w >> 8) & 3, jc = w >> 10;   // 4 col slices x 4 j-chunks
  int i0 = it * 16, rboff = (lane & 16) ? 8 : 0;
  v16bf aq[HEADS][2];
  float mh[HEADS][8], ih[HEADS][8];
  #pragma unroll
  for (int h = 0; h < HEADS; h++) {
    aq[h][0] = load_a16(qkvb + h * DHEAD, QKVW, i0, 0);
    aq[h][1] = load_a16(qkvb + h * DHEAD, QKVW, i0, 32);
    #pragma unroll
    for (int r = 0; r < 8; r++) {
      mh[h][r] = m1[h * NND + i0 + rboff + r];
      ih[h][r] = 1.0f / l1[h * NND + i0 + rboff + r];
    }
  }
  float fr[8], mr2[8], il2[8];
  #pragma unroll
  for (int r = 0; r < 8; r++) {
    fr[r]  = f1[i0 + rboff + r];
    mr2[r] = m2[i0 + rboff + r];
    il2[r] = 1.0f / l2[i0 + rboff + r];
  }
  v8f o0 = zero8(), o1 = zero8(), o2 = zero8(), o3 = zero8();
  bf16* myp = pt[wid];
  const bf16* wb = wht + (size_t)cs * 64 * NND;
  for (int j0 = jc * 1024; j0 < jc * 1024 + 1024; j0 += 32) {
    #pragma unroll
    for (int sub = 0; sub < 2; sub++) {
      int jj = j0 + sub * 16;
      v8f d = zero8();
      #pragma unroll
      for (int h = 0; h < HEADS; h++) {
        v16bf b0 = load_b16(qkvb + FOUT + h * DHEAD, QKVW, jj, 0);
        v16bf b1 = load_b16(qkvb + FOUT + h * DHEAD, QKVW, jj, 32);
        v8f s = zero8();
        s = wmma_bf16(aq[h][0], b0, s);
        s = wmma_bf16(aq[h][1], b1, s);
        #pragma unroll
        for (int r = 0; r < 8; r++) d[r] += __expf(s[r] * SCALE - mh[h][r]) * ih[h][r];
      }
      int j = jj + (lane & 15);
      float f2j = f2[j];
      int c = (lane & 15) + sub * 16;
      #pragma unroll
      for (int r = 0; r < 8; r++) {
        float e = fr[r] + f2j; e = e > 0.f ? e : ALPHA * e;
        int ad = adj[(size_t)(i0 + rboff + r) * NND + j];
        float t = ad > 0 ? (e + d[r]) : NEGBIG;
        float p = __expf(t - mr2[r]) * il2[r];
        myp[(rboff + r) * 32 + c] = (bf16)p;
      }
    }
    asm volatile("s_wait_dscnt 0" ::: "memory");
    v16bf pa = lds_a16(myp);
    o0 = wmma_bf16(pa, load_b16(wb, NND,  0, j0), o0);
    o1 = wmma_bf16(pa, load_b16(wb, NND, 16, j0), o1);
    o2 = wmma_bf16(pa, load_b16(wb, NND, 32, j0), o2);
    o3 = wmma_bf16(pa, load_b16(wb, NND, 48, j0), o3);
  }
  int cw = lane & 15;
  #pragma unroll
  for (int r = 0; r < 8; r++) {
    size_t row = (size_t)(i0 + rboff + r) * FOUT + cs * 64;
    atomicAdd(&hp[row +  0 + cw], o0[r]);
    atomicAdd(&hp[row + 16 + cw], o1[r]);
    atomicAdd(&hp[row + 32 + cw], o2[r]);
    atomicAdd(&hp[row + 48 + cw], o3[r]);
  }
}

__global__ void k_final(const float* hp, const float* wh0s, float* out) {
  int i = blockIdx.x * 256 + threadIdx.x;
  float x = hp[i] + wh0s[i];
  out[i] = x > 0.f ? x : (__expf(x) - 1.f);
}

// ---------------- host orchestration --------------------------------------

extern "C" void kernel_launch(void* const* d_in, const int* in_sizes, int n_in,
                              void* d_out, int out_size, void* d_ws, size_t ws_size,
                              hipStream_t stream) {
  const float* h      = (const float*)d_in[0];
  const int*   adj    = (const int*)  d_in[1];
  const float* W      = (const float*)d_in[2];
  const float* Wl_w   = (const float*)d_in[3];
  const float* Wl_b   = (const float*)d_in[4];
  const float* Wqkv_w = (const float*)d_in[5];
  const float* Wqkv_b = (const float*)d_in[6];
  const float* a      = (const float*)d_in[7];
  float* out = (float*)d_out;

  char* cur = (char*)d_ws;
  auto alloc = [&](size_t b) -> void* {
    void* p = cur; cur += ((b + 255) & ~size_t(255)); return p;
  };
  bf16*  hb   = (bf16*) alloc((size_t)NND * FIN * 2);
  bf16*  wt   = (bf16*) alloc((size_t)FOUT * FIN * 2);
  bf16*  wlb  = (bf16*) alloc((size_t)FOUT * FOUT * 2);
  bf16*  wqb  = (bf16*) alloc((size_t)QKVW * FOUT * 2);
  float* wh1  = (float*)alloc((size_t)NND * FOUT * 4);
  bf16*  wh1b = (bf16*) alloc((size_t)NND * FOUT * 2);
  float* wh   = (float*)alloc((size_t)NND * FOUT * 4);
  bf16*  whb  = (bf16*) alloc((size_t)NND * FOUT * 2);
  bf16*  wht  = (bf16*) alloc((size_t)FOUT * NND * 2);
  float* qkv  = (float*)alloc((size_t)NND * QKVW * 4);
  bf16*  qkvb = (bf16*) alloc((size_t)NND * QKVW * 2);
  bf16*  vt   = (bf16*) alloc((size_t)FOUT * NND * 2);
  float* m1   = (float*)alloc((size_t)HEADS * NND * 4);
  float* l1   = (float*)alloc((size_t)HEADS * NND * 4);
  float* wh0  = (float*)alloc((size_t)NND * FOUT * 4);
  float* wh0s = (float*)alloc((size_t)NND * FOUT * 4);
  float* f1v  = (float*)alloc((size_t)NND * 4);
  float* f2v  = (float*)alloc((size_t)NND * 4);
  float* m2p  = (float*)alloc((size_t)8 * NND * 4);
  float* l2p  = (float*)alloc((size_t)8 * NND * 4);
  float* m2   = (float*)alloc((size_t)NND * 4);
  float* l2   = (float*)alloc((size_t)NND * 4);
  float* hp   = (float*)alloc((size_t)NND * FOUT * 4);

  // 1. conversions / transposes
  k_cvt <<<8192, 256, 0, stream>>>(h, hb, NND * FIN);
  k_tcvt<<< 512, 256, 0, stream>>>(W, wt, FIN, FOUT);
  k_cvt <<< 256, 256, 0, stream>>>(Wl_w, wlb, FOUT * FOUT);
  k_cvt <<< 768, 256, 0, stream>>>(Wqkv_w, wqb, QKVW * FOUT);

  // 2. projection chain (WMMA GEMMs)
  k_gemm<<<1024, 128, 0, stream>>>(hb, wt, nullptr, wh1, NND, FOUT / 16, FIN);
  k_cvt <<<4096, 256, 0, stream>>>(wh1, wh1b, NND * FOUT);
  k_gemm<<<1024, 128, 0, stream>>>(wh1b, wlb, Wl_b, wh, NND, FOUT / 16, FOUT);
  k_cvt <<<4096, 256, 0, stream>>>(wh, whb, NND * FOUT);
  k_tcvt<<<4096, 256, 0, stream>>>(wh, wht, NND, FOUT);
  k_gemm<<<3072, 128, 0, stream>>>(whb, wqb, Wqkv_b, qkv, NND, QKVW / 16, FOUT);
  k_cvt <<<12288, 256, 0, stream>>>(qkv, qkvb, NND * QKVW);
  k_vtk <<<4096, 256, 0, stream>>>(qkv, vt);

  // 3. per-head attention stats + Wh0 = dots@V + row softmax
  k_stats1<<<256, 128, 0, stream>>>(qkvb, m1, l1);
  k_wh0   <<<256, 128, 0, stream>>>(qkvb, vt, m1, l1, wh0);
  k_softmax256<<<NND, 256, 0, stream>>>(wh0, wh0s);

  // 4. e-vectors
  k_gemv2<<<512, 256, 0, stream>>>(wh, a, f1v, f2v);

  // 5. masked graph softmax stats (j-split) + combine
  k_stats2<<<512, 128, 0, stream>>>(qkvb, m1, l1, f1v, f2v, adj, m2p, l2p);
  k_comb2 <<< 16, 256, 0, stream>>>(m2p, l2p, m2, l2);

  // 6. h' = attention @ Wh (j-split, f32 atomics), then elu(h' + Wh0s)
  hipMemsetAsync(hp, 0, (size_t)NND * FOUT * 4, stream);
  k_passB<<<1024, 128, 0, stream>>>(qkvb, m1, l1, f1v, f2v, adj, m2, l2, wht, hp);
  k_final<<<4096, 256, 0, stream>>>(hp, wh0s, out);
}
